// DenseCGPrior_6708738916913
// MI455X (gfx1250) — matrix operand order repaired and checked by reading, not compiled, not run on hardware
//
#include <hip/hip_runtime.h>

#define DINLINE __device__ __forceinline__

typedef __attribute__((ext_vector_type(16))) _Float16 v16h;
typedef __attribute__((ext_vector_type(8)))  float    v8f;

constexpr int   Bc    = 4;
constexpr int   Nc    = 128;
constexpr int   Fc    = 128;
constexpr int   Lc    = 3;
constexpr int   NRBF  = 20;
constexpr int   F3    = 3 * Fc;     // 384
constexpr int   NODES = Bc * Nc;    // 512
constexpr float EPSc  = 0.001f;
constexpr float PIc   = 3.14159265358979323846f;
constexpr float CUT   = 5.0f;

// msg kernel tiling: i-tile of 8 nodes, j range split in 2 slices of 64
constexpr int ITILE = 8;
constexpr int JS    = 2;            // j slices
constexpr int JTPB  = 4;            // j-tiles (of 16) per block

// ---------------------------------------------------------------------------
// WMMA helpers (CDNA5 wave32, f16 inputs, f32 accumulate, 16x16x32)
// ---------------------------------------------------------------------------
DINLINE v8f wmma16(v16h a, v16h b, v8f c) {
  return __builtin_amdgcn_wmma_f32_16x16x32_f16(
      /*neg_a=*/false, a, /*neg_b=*/false, b,
      /*c_mod=*/(short)0, c, /*reuse_a=*/false, /*reuse_b=*/false);
}

// A fragment (16x32 f16, M x K) from row-major f16 memory.
// `row` points at A[m][k0] for this lane's row m = lane&15; h = lane>>4.
// ISA 16-bit A layout: VGPR v holds K = (v&3)*2 + (v>>2)*16 + h*8, +1.
DINLINE v16h load_a_frag(const _Float16* row, int h) {
  v16h a;
#pragma unroll
  for (int v = 0; v < 8; ++v) {
    int kb = ((v & 3) << 1) + ((v >> 2) << 4) + (h << 3);
    a[2 * v]     = row[kb];
    a[2 * v + 1] = row[kb + 1];
  }
  return a;
}

// B fragment (32x16 f16, K x N), pre-swizzled in global memory:
// frag[(kt*NT+nt)*32 + lane][e], element e holds B[(e&7) + (lane>>4)*8 + (e>>3)*16][lane&15].
DINLINE v16h load_b_frag(const _Float16* base, int kt, int nt, int NT, int lane) {
  return *(const v16h*)(base + (((size_t)(kt * NT + nt) * 32) + lane) * 16);
}

DINLINE float silu_f(float x) { return x / (1.0f + __expf(-x)); }

// Async global->LDS copy of 16B (tracked with ASYNCcnt), gfx1250 path.
DINLINE void async_ld_b128(unsigned int lds_off, const float* g) {
  asm volatile("global_load_async_to_lds_b128 %0, %1, off"
               :: "v"(lds_off), "v"(g) : "memory");
}
DINLINE void wait_async0() {
  asm volatile("s_wait_asynccnt 0x0" ::: "memory");
}

// ---------------------------------------------------------------------------
// Weight swizzle: row-major f32 W (Krows x Ncols) -> f16 B-fragment layout.
// Optional extra row (used to append rbf bias as feature row `extra_idx`).
// ---------------------------------------------------------------------------
__global__ void swizzle_b(const float* __restrict__ W,
                          const float* __restrict__ extra_row, int extra_idx,
                          int Krows, int Ncols, _Float16* __restrict__ dst) {
  int kt = blockIdx.x, nt = blockIdx.y, NT = (int)gridDim.y;
  int lane = threadIdx.x;
  int h = lane >> 4, nl = lane & 15;
  int col = nt * 16 + nl;
  v16h frag;
#pragma unroll
  for (int e = 0; e < 16; ++e) {
    int k = kt * 32 + (e & 7) + h * 8 + ((e >> 3) << 4);
    float val = 0.0f;
    if (k < Krows)           val = W[(size_t)k * Ncols + col];
    else if (k == extra_idx) val = extra_row[col];
    frag[e] = (_Float16)val;
  }
  *(v16h*)(dst + (((size_t)(kt * NT + nt) * 32) + lane) * 16) = frag;
}

// ---------------------------------------------------------------------------
// Small utility kernels
// ---------------------------------------------------------------------------
__global__ void copyf(float* __restrict__ dst, const float* __restrict__ src, int n) {
  int i = blockIdx.x * blockDim.x + threadIdx.x;
  if (i < n) dst[i] = src[i];
}
__global__ void fillz(float* __restrict__ dst, int n) {
  int i = blockIdx.x * blockDim.x + threadIdx.x;
  if (i < n) dst[i] = 0.0f;
}
// o = a + p0 + p1  (deterministic cross-block reduction of j-slice partials)
__global__ void reduce3(const float* __restrict__ a, const float* __restrict__ p0,
                        const float* __restrict__ p1, float* __restrict__ o, int n) {
  int i = blockIdx.x * blockDim.x + threadIdx.x;
  if (i < n) o[i] = a[i] + p0[i] + p1[i];
}
__global__ void deg_kernel(const float* __restrict__ adj, float* __restrict__ dis) {
  int bi = blockIdx.x * blockDim.x + threadIdx.x;
  if (bi < NODES) {
    float d = 0.0f;
    for (int j = 0; j < Nc; ++j) d += adj[(size_t)bi * Nc + j];
    dis[bi] = sqrtf(1.0f / d + EPSc);
  }
}

// ---------------------------------------------------------------------------
// Edge precompute: coef = ew*env, unit vectors, and RBF features written
// DIRECTLY in WMMA A-fragment layout (feature 20 == 1.0 carries the bias row).
// ---------------------------------------------------------------------------
__global__ __launch_bounds__(256) void edge_kernel(
    const float* __restrict__ adj, const float* __restrict__ xyz,
    const float* __restrict__ dis,
    float* __restrict__ coef, float* __restrict__ unitv,
    _Float16* __restrict__ rbfA) {
  int bi = blockIdx.x;
  int b = bi >> 7, i = bi & (Nc - 1);
  int t = threadIdx.x;
  int wv = t >> 5, lane = t & 31, h = lane >> 4, m = lane & 15;
  int j = wv * 16 + m;

  float rx = xyz[((size_t)b * Nc + j) * 3 + 0] - xyz[((size_t)b * Nc + i) * 3 + 0];
  float ry = xyz[((size_t)b * Nc + j) * 3 + 1] - xyz[((size_t)b * Nc + i) * 3 + 1];
  float rz = xyz[((size_t)b * Nc + j) * 3 + 2] - xyz[((size_t)b * Nc + i) * 3 + 2];
  float d  = sqrtf(rx * rx + ry * ry + rz * rz + EPSc);
  float env = (d < CUT) ? 0.5f * (__cosf(PIc * d / CUT) + 1.0f) : 0.0f;
  float aij = adj[(size_t)bi * Nc + j];
  float ew  = (aij > 0.0f ? 1.0f : 0.0f) * dis[b * Nc + i] * dis[b * Nc + j];

  if (h == 0) {
    coef[(size_t)bi * Nc + j] = ew * env;
    unitv[((size_t)bi * Nc + j) * 3 + 0] = rx / d;
    unitv[((size_t)bi * Nc + j) * 3 + 1] = ry / d;
    unitv[((size_t)bi * Nc + j) * 3 + 2] = rz / d;
  }

  v16h frag;
#pragma unroll
  for (int e = 0; e < 16; ++e) {
    int vv = e >> 1;
    int k  = ((vv & 3) << 1) + ((vv >> 2) << 4) + (h << 3) + (e & 1);
    float val;
    if (k < NRBF) val = __sinf(d * (float)(k + 1) * (PIc / CUT)) / d;
    else          val = (k == NRBF) ? 1.0f : 0.0f;
    frag[e] = (_Float16)val;
  }
  *(v16h*)(rbfA + ((size_t)(bi * 8 + wv) * 32 + lane) * 16) = frag;
}

// ---------------------------------------------------------------------------
// Generic 2-layer node MLP (templated -> fully unrolled GEMMs):
// out = act2( act1(X @ W1 + b1) @ W2 + b2 ); X:(NODES x 128), hidden 128.
// ---------------------------------------------------------------------------
template <int NT2, int ACT1, int ACT2>
__global__ __launch_bounds__(128) void node_mlp(
    const float* __restrict__ X,
    const _Float16* __restrict__ W1f, const float* __restrict__ b1,
    const _Float16* __restrict__ W2f, const float* __restrict__ b2,
    float* __restrict__ out) {
  __shared__ _Float16 xs[16 * Fc];
  __shared__ _Float16 hs[16 * Fc];
  int rt = blockIdx.x;
  int t = threadIdx.x;
  int wv = t >> 5, lane = t & 31, h = lane >> 4, nl = lane & 15;

  for (int idx = t; idx < 16 * Fc; idx += 128)
    xs[idx] = (_Float16)X[(size_t)(rt * 16 + (idx >> 7)) * Fc + (idx & 127)];
  __syncthreads();

#pragma unroll
  for (int q = 0; q < 2; ++q) {            // hidden GEMM: 8 n-tiles / 4 waves
    int nt = wv * 2 + q;
    v8f acc = {};
#pragma unroll
    for (int kt = 0; kt < 4; ++kt) {
      v16h a  = load_a_frag(&xs[nl * Fc + kt * 32], h);
      v16h bf = load_b_frag(W1f, kt, nt, 8, lane);
      acc = wmma16(a, bf, acc);
    }
    int n = nt * 16 + nl;
    float bias = b1[n];
#pragma unroll
    for (int r = 0; r < 8; ++r) {
      int mm = r + h * 8;
      float x = acc[r] + bias;
      x = (ACT1 == 0) ? silu_f(x) : tanhf(x);
      hs[mm * Fc + n] = (_Float16)x;
    }
  }
  __syncthreads();

#pragma unroll
  for (int nt = 0; nt < NT2 / 4; ++nt) {   // output GEMM, fully unrolled
    int ntw = nt * 4 + wv;
    v8f acc = {};
#pragma unroll
    for (int kt = 0; kt < 4; ++kt) {
      v16h a  = load_a_frag(&hs[nl * Fc + kt * 32], h);
      v16h bf = load_b_frag(W2f, kt, ntw, NT2, lane);
      acc = wmma16(a, bf, acc);
    }
    int n = ntw * 16 + nl;
    float bias = b2[n];
#pragma unroll
    for (int r = 0; r < 8; ++r) {
      int mm = r + h * 8;
      float x = acc[r] + bias;
      if (ACT2 == 1) x = 1e-9f + __expf(0.5f * x);
      out[(size_t)(rt * 16 + mm) * (NT2 * 16) + n] = x;
    }
  }
}

// ---------------------------------------------------------------------------
// Fused message kernel, i-tiled + j-sliced.
// Block = (b, i-tile of 8, j-slice of 64). 256 threads = 8 waves.
// Per j-tile of 16: async-stage phi (f32, ASYNCcnt path) + v (f16) once,
// reuse across 8 i-subtiles. Per (i-sub, j-tile): one WMMA per segment
// (m0/m1/m2); accumulator (rows=j, cols=feature) is scaled by env*ew*phi and
// reduced over j in-register into LDS partials. Wave w owns features
// [16w,16w+16). Writes per-slice partial ds/dv (deterministically reduced
// by reduce3). The (B,N,N,3F) edge tensor is never materialized.
// ---------------------------------------------------------------------------
__global__ __launch_bounds__(256) void msg_kernel(
    const float* __restrict__ v_in, const float* __restrict__ phi,
    const _Float16* __restrict__ rbfA, const _Float16* __restrict__ rbfWf,
    const float* __restrict__ coef, const float* __restrict__ unitv,
    float* __restrict__ s_part, float* __restrict__ v_part) {
  __shared__ alignas(16) float    phi_s[16 * F3];        // 24 KB (async dst)
  __shared__ _Float16             v_h[16 * F3];          // 12 KB
  __shared__ float                ds_acc[ITILE * Fc];    //  4 KB
  __shared__ float                dv_acc[ITILE * Fc * 3];// 12 KB
  __shared__ float                coef_s[ITILE * 16];
  __shared__ float                unit_s[ITILE * 16 * 3];

  int bx = blockIdx.x;
  int js = bx & (JS - 1);
  int it = (bx >> 1) & 15;
  int b  = bx >> 5;
  int i0 = it * ITILE;
  int t  = threadIdx.x;
  int wv = t >> 5, lane = t & 31, h = lane >> 4, nl = lane & 15;
  int f  = wv * 16 + nl;                    // feature column owned by this lane

  for (int idx = t; idx < ITILE * Fc; idx += 256)     ds_acc[idx] = 0.0f;
  for (int idx = t; idx < ITILE * Fc * 3; idx += 256) dv_acc[idx] = 0.0f;

  // rbf-weight B fragments are j-tile invariant: keep in VGPRs for the block
  v16h bf0 = load_b_frag(rbfWf, 0, wv,      24, lane);
  v16h bf1 = load_b_frag(rbfWf, 0, wv + 8,  24, lane);
  v16h bf2 = load_b_frag(rbfWf, 0, wv + 16, 24, lane);

  unsigned int phi_lds = (unsigned int)(size_t)(&phi_s[0]);

  for (int q = 0; q < JTPB; ++q) {
    int jt = js * JTPB + q;
    int j0 = jt * 16;
    __syncthreads();                        // previous tile fully consumed
    // ---- async-stage phi tile (16 x 384 f32, contiguous rows) ----
    const float* gphi = &phi[((size_t)b * Nc + j0) * F3];
    for (int idx = t; idx < (16 * F3) / 4; idx += 256)
      async_ld_b128(phi_lds + idx * 16, gphi + idx * 4);
    // ---- stage v tile as f16, coef/unit per (i,j) ----
    for (int idx = t; idx < 16 * F3; idx += 256)
      v_h[idx] = (_Float16)v_in[((size_t)b * Nc + j0) * F3 + idx];
    for (int idx = t; idx < ITILE * 16; idx += 256)
      coef_s[idx] = coef[((size_t)(b * Nc + i0 + (idx >> 4))) * Nc + j0 + (idx & 15)];
    for (int idx = t; idx < ITILE * 16 * 3; idx += 256) {
      int isub = idx / 48, rr = idx % 48, jj = rr / 3, c = rr % 3;
      unit_s[idx] = unitv[((size_t)(b * Nc + i0 + isub) * Nc + j0 + jj) * 3 + c];
    }
    // prefetch next j-tile's A fragments while we compute on this one
    if (q < JTPB - 1) {
      const _Float16* nxt =
          rbfA + ((size_t)((b * Nc + i0) * 8 + jt + 1) * 32 + lane) * 16;
      __builtin_prefetch(nxt, 0, 0);
    }
    wait_async0();                          // this wave's async copies done
    __syncthreads();                        // all waves' staging visible

#pragma unroll
    for (int isub = 0; isub < ITILE; ++isub) {
      v16h a = *(const v16h*)(
          rbfA + ((size_t)((b * Nc + i0 + isub) * 8 + jt) * 32 + lane) * 16);
      v8f c0 = {}, c1 = {}, c2 = {};
      c0 = wmma16(a, bf0, c0);              // m0 block: cols f
      c1 = wmma16(a, bf1, c1);              // m1 block: cols 128+f
      c2 = wmma16(a, bf2, c2);              // m2 block: cols 256+f
      float sa = 0, d0 = 0, d1 = 0, d2 = 0;
#pragma unroll
      for (int r = 0; r < 8; ++r) {
        int jj = r + h * 8;
        float cf = coef_s[isub * 16 + jj];
        float m0 = c0[r] * cf * phi_s[jj * F3 + f];
        float m1 = c1[r] * cf * phi_s[jj * F3 + Fc + f];
        float m2 = c2[r] * cf * phi_s[jj * F3 + 2 * Fc + f];
        sa += m1;
        float ux = unit_s[(isub * 16 + jj) * 3 + 0];
        float uy = unit_s[(isub * 16 + jj) * 3 + 1];
        float uz = unit_s[(isub * 16 + jj) * 3 + 2];
        d0 += m0 * (float)v_h[jj * F3 + f * 3 + 0] + m2 * ux;
        d1 += m0 * (float)v_h[jj * F3 + f * 3 + 1] + m2 * uy;
        d2 += m0 * (float)v_h[jj * F3 + f * 3 + 2] + m2 * uz;
      }
      sa += __shfl_xor(sa, 16, 32);
      d0 += __shfl_xor(d0, 16, 32);
      d1 += __shfl_xor(d1, 16, 32);
      d2 += __shfl_xor(d2, 16, 32);
      if (h == 0) {                         // wave-exclusive feature range
        ds_acc[isub * Fc + f] += sa;
        dv_acc[(isub * Fc + f) * 3 + 0] += d0;
        dv_acc[(isub * Fc + f) * 3 + 1] += d1;
        dv_acc[(isub * Fc + f) * 3 + 2] += d2;
      }
    }
  }
  __syncthreads();
  // ---- write contiguous per-slice partials ----
  float* sp = s_part + (size_t)js * NODES * Fc + (size_t)(b * Nc + i0) * Fc;
  float* vp = v_part + (size_t)js * NODES * F3 + (size_t)(b * Nc + i0) * F3;
  for (int idx = t; idx < ITILE * Fc; idx += 256)     sp[idx] = ds_acc[idx];
  for (int idx = t; idx < ITILE * Fc * 3; idx += 256) vp[idx] = dv_acc[idx];
}

// ---------------------------------------------------------------------------
// Update step (node-local, in place). Block = 16-node tile, 256 thr = 8 waves.
// ---------------------------------------------------------------------------
__global__ __launch_bounds__(256) void upd_kernel(
    float* __restrict__ s, float* __restrict__ v,
    const _Float16* __restrict__ Wuf, const _Float16* __restrict__ Wvf,
    const _Float16* __restrict__ Ws1f, const float* __restrict__ bs1,
    const _Float16* __restrict__ Ws2f, const float* __restrict__ bs2) {
  __shared__ _Float16 va[3][16 * Fc];       // v as A-matrix per channel
  __shared__ _Float16 uvh[16 * Fc * 3];     // u_v (f16)
  __shared__ float    dotb[16 * Fc];        // sum_c u_v*v_v
  __shared__ float    nsq[16 * Fc];         // sum_c v_v^2
  __shared__ _Float16 sstack[16 * 2 * Fc];  // [s, |v_v|] f16
  __shared__ _Float16 hid[16 * Fc];

  int rt = blockIdx.x;
  int t = threadIdx.x;
  int wv = t >> 5, lane = t & 31, h = lane >> 4, nl = lane & 15;

  for (int idx = t; idx < 16 * F3; idx += 256) {
    int mm = idx / F3, fc = idx % F3, f = fc / 3, c = fc % 3;
    va[c][mm * Fc + f] = (_Float16)v[(size_t)(rt * 16 + mm) * F3 + fc];
  }
  __syncthreads();

  float dotp[8] = {}, np[8] = {};
  float uvals[3][8];
#pragma unroll
  for (int c = 0; c < 3; ++c) {
    v8f au = {}, av = {};
#pragma unroll
    for (int kt = 0; kt < 4; ++kt) {
      v16h a = load_a_frag(&va[c][nl * Fc + kt * 32], h);
      au = wmma16(a, load_b_frag(Wuf, kt, wv, 8, lane), au);
      av = wmma16(a, load_b_frag(Wvf, kt, wv, 8, lane), av);
    }
#pragma unroll
    for (int r = 0; r < 8; ++r) {
      float u = au[r], w2 = av[r];
      dotp[r] += u * w2;
      np[r]   += w2 * w2;
      uvals[c][r] = u;
    }
  }
  int g = wv * 16 + nl;
#pragma unroll
  for (int r = 0; r < 8; ++r) {
    int mm = r + h * 8;
    dotb[mm * Fc + g] = dotp[r];
    nsq[mm * Fc + g]  = np[r];
#pragma unroll
    for (int c = 0; c < 3; ++c) uvh[(mm * Fc + g) * 3 + c] = (_Float16)uvals[c][r];
  }
  __syncthreads();

  for (int idx = t; idx < 16 * Fc; idx += 256) {
    int mm = idx >> 7, f = idx & 127;
    sstack[mm * 256 + f]      = (_Float16)s[(size_t)(rt * 16 + mm) * Fc + f];
    sstack[mm * 256 + Fc + f] = (_Float16)sqrtf(nsq[idx] + EPSc);
  }
  __syncthreads();

  {                                          // GEMM Ws1: K=256
    v8f acc = {};
#pragma unroll
    for (int kt = 0; kt < 8; ++kt) {
      v16h a = load_a_frag(&sstack[nl * 256 + kt * 32], h);
      acc = wmma16(a, load_b_frag(Ws1f, kt, wv, 8, lane), acc);
    }
    int n = wv * 16 + nl;
    float bias = bs1[n];
#pragma unroll
    for (int r = 0; r < 8; ++r) {
      int mm = r + h * 8;
      hid[mm * Fc + n] = (_Float16)silu_f(acc[r] + bias);
    }
  }
  __syncthreads();

  v8f cvv = {}, csv = {}, css = {};          // a_vv, a_sv, a_ss for f-range
#pragma unroll
  for (int kt = 0; kt < 4; ++kt) {
    v16h a = load_a_frag(&hid[nl * Fc + kt * 32], h);
    cvv = wmma16(a, load_b_frag(Ws2f, kt, wv,      24, lane), cvv);
    csv = wmma16(a, load_b_frag(Ws2f, kt, wv + 8,  24, lane), csv);
    css = wmma16(a, load_b_frag(Ws2f, kt, wv + 16, 24, lane), css);
  }
  int f = wv * 16 + nl;
#pragma unroll
  for (int r = 0; r < 8; ++r) {
    int mm = r + h * 8;
    int node = rt * 16 + mm;
    float avv = cvv[r] + bs2[f];
    float asv = csv[r] + bs2[Fc + f];
    float ass = css[r] + bs2[2 * Fc + f];
    s[(size_t)node * Fc + f] += asv * dotb[mm * Fc + f] + ass;
#pragma unroll
    for (int c = 0; c < 3; ++c) {
      float uv = (float)uvh[(mm * Fc + f) * 3 + c];
      v[(size_t)node * F3 + f * 3 + c] += uv * avv;
    }
  }
}

// ---------------------------------------------------------------------------
// Host driver
// ---------------------------------------------------------------------------
extern "C" void kernel_launch(void* const* d_in, const int* in_sizes, int n_in,
                              void* d_out, int out_size, void* d_ws, size_t ws_size,
                              hipStream_t stream) {
  (void)in_sizes; (void)n_in; (void)out_size; (void)ws_size;
  const float* H       = (const float*)d_in[0];
  const float* adj     = (const float*)d_in[1];
  const float* xyz     = (const float*)d_in[2];
  const float* msg_W1  = (const float*)d_in[3];
  const float* msg_b1  = (const float*)d_in[4];
  const float* msg_W2  = (const float*)d_in[5];
  const float* msg_b2  = (const float*)d_in[6];
  const float* rbf_W   = (const float*)d_in[7];
  const float* rbf_b   = (const float*)d_in[8];
  const float* upd_Wu  = (const float*)d_in[9];
  const float* upd_Wv  = (const float*)d_in[10];
  const float* upd_Ws1 = (const float*)d_in[11];
  const float* upd_bs1 = (const float*)d_in[12];
  const float* upd_Ws2 = (const float*)d_in[13];
  const float* upd_bs2 = (const float*)d_in[14];
  const float* mu_W1   = (const float*)d_in[15];
  const float* mu_b1   = (const float*)d_in[16];
  const float* mu_W2   = (const float*)d_in[17];
  const float* mu_b2   = (const float*)d_in[18];
  const float* sig_W1  = (const float*)d_in[19];
  const float* sig_b1  = (const float*)d_in[20];
  const float* sig_W2  = (const float*)d_in[21];
  const float* sig_b2  = (const float*)d_in[22];
  float* out = (float*)d_out;

  // ---- workspace carve (f32 region, then f16 region) ----
  float* w = (float*)d_ws;
  float* s0     = w; w += NODES * Fc;
  float* s1     = w; w += NODES * Fc;
  float* v0     = w; w += NODES * F3;
  float* v1     = w; w += NODES * F3;
  float* phi    = w; w += NODES * F3;
  float* coef   = w; w += NODES * Nc;
  float* unitv  = w; w += NODES * Nc * 3;
  float* dis    = w; w += NODES;
  float* s_part = w; w += (size_t)JS * NODES * Fc;
  float* v_part = w; w += (size_t)JS * NODES * F3;
  _Float16* hb = (_Float16*)w;
  _Float16* rbfA = hb; hb += (size_t)NODES * 8 * 512;   // A-frags for rbf GEMM

  auto take = [&](size_t n) { _Float16* p = hb; hb += n; return p; };
  _Float16 *wf_msgW1[Lc], *wf_msgW2[Lc], *wf_rbfW[Lc], *wf_Wu[Lc], *wf_Wv[Lc],
           *wf_Ws1[Lc], *wf_Ws2[Lc];
  for (int l = 0; l < Lc; ++l) {
    wf_msgW1[l] = take((size_t)4 * 8 * 512);
    wf_msgW2[l] = take((size_t)4 * 24 * 512);
    wf_rbfW[l]  = take((size_t)1 * 24 * 512);
    wf_Wu[l]    = take((size_t)4 * 8 * 512);
    wf_Wv[l]    = take((size_t)4 * 8 * 512);
    wf_Ws1[l]   = take((size_t)8 * 8 * 512);
    wf_Ws2[l]   = take((size_t)4 * 24 * 512);
  }
  _Float16* wf_mu1 = take((size_t)4 * 8 * 512);
  _Float16* wf_mu2 = take((size_t)4 * 8 * 512);
  _Float16* wf_sg1 = take((size_t)4 * 8 * 512);
  _Float16* wf_sg2 = take((size_t)4 * 8 * 512);

  // ---- convert weights to pre-swizzled f16 B fragments ----
  for (int l = 0; l < Lc; ++l) {
    swizzle_b<<<dim3(4, 8),  32, 0, stream>>>(msg_W1  + (size_t)l * Fc * Fc,     msg_W1, -1, Fc,     Fc, wf_msgW1[l]);
    swizzle_b<<<dim3(4, 24), 32, 0, stream>>>(msg_W2  + (size_t)l * Fc * F3,     msg_W2, -1, Fc,     F3, wf_msgW2[l]);
    swizzle_b<<<dim3(1, 24), 32, 0, stream>>>(rbf_W   + (size_t)l * NRBF * F3,
                                              rbf_b   + (size_t)l * F3, NRBF, NRBF, F3, wf_rbfW[l]);
    swizzle_b<<<dim3(4, 8),  32, 0, stream>>>(upd_Wu  + (size_t)l * Fc * Fc,     upd_Wu, -1, Fc,     Fc, wf_Wu[l]);
    swizzle_b<<<dim3(4, 8),  32, 0, stream>>>(upd_Wv  + (size_t)l * Fc * Fc,     upd_Wv, -1, Fc,     Fc, wf_Wv[l]);
    swizzle_b<<<dim3(8, 8),  32, 0, stream>>>(upd_Ws1 + (size_t)l * 2 * Fc * Fc, upd_Ws1, -1, 2 * Fc, Fc, wf_Ws1[l]);
    swizzle_b<<<dim3(4, 24), 32, 0, stream>>>(upd_Ws2 + (size_t)l * Fc * F3,     upd_Ws2, -1, Fc,     F3, wf_Ws2[l]);
  }
  swizzle_b<<<dim3(4, 8), 32, 0, stream>>>(mu_W1,  mu_W1,  -1, Fc, Fc, wf_mu1);
  swizzle_b<<<dim3(4, 8), 32, 0, stream>>>(mu_W2,  mu_W2,  -1, Fc, Fc, wf_mu2);
  swizzle_b<<<dim3(4, 8), 32, 0, stream>>>(sig_W1, sig_W1, -1, Fc, Fc, wf_sg1);
  swizzle_b<<<dim3(4, 8), 32, 0, stream>>>(sig_W2, sig_W2, -1, Fc, Fc, wf_sg2);

  // ---- state init + edge precompute ----
  copyf<<<(NODES * Fc) / 256, 256, 0, stream>>>(s0, H, NODES * Fc);
  fillz<<<(NODES * F3) / 256, 256, 0, stream>>>(v0, NODES * F3);
  deg_kernel<<<(NODES + 255) / 256, 256, 0, stream>>>(adj, dis);
  edge_kernel<<<NODES, 256, 0, stream>>>(adj, xyz, dis, coef, unitv, rbfA);

  // ---- L layers: phi -> fused message (partials) -> reduce -> update ----
  float *sp = s0, *vp = v0, *sq = s1, *vq = v1;
  for (int l = 0; l < Lc; ++l) {
    node_mlp<24, 0, 0><<<NODES / 16, 128, 0, stream>>>(
        sp, wf_msgW1[l], msg_b1 + (size_t)l * Fc,
        wf_msgW2[l], msg_b2 + (size_t)l * F3, phi);
    msg_kernel<<<Bc * (Nc / ITILE) * JS, 256, 0, stream>>>(
        vp, phi, rbfA, wf_rbfW[l], coef, unitv, s_part, v_part);
    reduce3<<<(NODES * Fc) / 256, 256, 0, stream>>>(
        sp, s_part, s_part + (size_t)NODES * Fc, sq, NODES * Fc);
    reduce3<<<(NODES * F3) / 256, 256, 0, stream>>>(
        vp, v_part, v_part + (size_t)NODES * F3, vq, NODES * F3);
    upd_kernel<<<NODES / 16, 256, 0, stream>>>(
        sq, vq, wf_Wu[l], wf_Wv[l], wf_Ws1[l], upd_bs1 + (size_t)l * Fc,
        wf_Ws2[l], upd_bs2 + (size_t)l * F3);
    float* ts = sp; sp = sq; sq = ts;
    float* tv = vp; vp = vq; vq = tv;
  }

  // ---- output heads: mu then sigma (1e-9 + exp(x/2)) ----
  node_mlp<8, 1, 0><<<NODES / 16, 128, 0, stream>>>(
      sp, wf_mu1, mu_b1, wf_mu2, mu_b2, out);
  node_mlp<8, 1, 1><<<NODES / 16, 128, 0, stream>>>(
      sp, wf_sg1, sig_b1, wf_sg2, sig_b2, out + (size_t)NODES * Fc);
}
